// GraphAttentionLayer_47184510714010
// MI455X (gfx1250) — matrix operand order repaired and checked by reading, not compiled, not run on hardware
//
#include <hip/hip_runtime.h>
#include <hip/hip_fp16.h>

typedef __attribute__((ext_vector_type(16))) _Float16 v16h;
typedef __attribute__((ext_vector_type(8)))  float    v8f;
typedef __attribute__((ext_vector_type(4)))  int      v4i;

#define GAT_B 4
#define GAT_N 4096
#define GAT_F 128
#define GAT_ALPHA 0.2f

// ---- workspace layout (bytes) ----
#define WS_WH_F32   0               // B*N*F f32   = 8 MB
#define WS_WH_F16   8388608         // B*N*F f16   = 4 MB
#define WS_W_F16    12582912        // F*F   f16   = 32 KB
#define WS_S        12615680        // B*N   f32   = 64 KB
#define WS_UW       12681216        // B*N   f32x2 = 128 KB  (u=exp(t), w=exp(0.2t))
#define WS_TMAX     12812288        // B     f32

#if defined(__AMDGCN__) && __has_builtin(__builtin_amdgcn_global_load_async_to_lds_b128)
#define GAT_ASYNC_STAGE 1
#else
#define GAT_ASYNC_STAGE 0
#endif

__device__ __forceinline__ void gat_wait_async0() {
#if defined(__AMDGCN__)
#if __has_builtin(__builtin_amdgcn_s_wait_asynccnt)
  __builtin_amdgcn_s_wait_asynccnt(0);
#else
  asm volatile("s_wait_asynccnt 0x0" ::: "memory");
#endif
#endif
}

// ---------------------------------------------------------------------------
// Kernel 0: W (128x128 f32) -> f16 row-major
// ---------------------------------------------------------------------------
__global__ void gat_wprep_kernel(const float* __restrict__ W,
                                 _Float16* __restrict__ Wf16) {
  int i = blockIdx.x * blockDim.x + threadIdx.x;
  if (i < GAT_F * GAT_F) Wf16[i] = (_Float16)W[i];
}

// ---------------------------------------------------------------------------
// Kernel 1: Wh = h @ W  via v_wmma_f32_16x16x32_f16.
// One wave per 16-row block; 8 accumulators cover all 128 output features.
// ---------------------------------------------------------------------------
__global__ __launch_bounds__(128) void gat_gemm1_kernel(
    const float* __restrict__ h, const _Float16* __restrict__ Wf16,
    float* __restrict__ Wh) {
  int tid  = threadIdx.x;
  int wave = tid >> 5;
  int lane = tid & 31;
  int m    = lane & 15;
  int lh   = lane >> 4;        // lane-half: 0 or 1
  int k0   = lh * 8;
  long rowbase = (long)blockIdx.x * 64 + (long)wave * 16;   // over B*N rows
  const float* hrow = h + (rowbase + m) * GAT_F;

  v8f acc[8] = {};
  for (int ks = 0; ks < 4; ++ks) {
    const float* hp = hrow + ks * 32 + k0;
    float4 a0 = *(const float4*)(hp);
    float4 a1 = *(const float4*)(hp + 4);
    float4 a2 = *(const float4*)(hp + 16);
    float4 a3 = *(const float4*)(hp + 20);
    v16h af;
    af[0]  = (_Float16)a0.x; af[1]  = (_Float16)a0.y;
    af[2]  = (_Float16)a0.z; af[3]  = (_Float16)a0.w;
    af[4]  = (_Float16)a1.x; af[5]  = (_Float16)a1.y;
    af[6]  = (_Float16)a1.z; af[7]  = (_Float16)a1.w;
    af[8]  = (_Float16)a2.x; af[9]  = (_Float16)a2.y;
    af[10] = (_Float16)a2.z; af[11] = (_Float16)a2.w;
    af[12] = (_Float16)a3.x; af[13] = (_Float16)a3.y;
    af[14] = (_Float16)a3.z; af[15] = (_Float16)a3.w;

    const _Float16* wrow = Wf16 + (ks * 32 + k0) * GAT_F + m;
#pragma unroll
    for (int ft = 0; ft < 8; ++ft) {
      v16h bf;
#pragma unroll
      for (int hh = 0; hh < 8; ++hh) {
        bf[hh]     = wrow[hh * GAT_F + ft * 16];
        bf[8 + hh] = wrow[(16 + hh) * GAT_F + ft * 16];
      }
      acc[ft] = __builtin_amdgcn_wmma_f32_16x16x32_f16(
          false, af, false, bf, (short)0, acc[ft], false, false);
    }
  }
#pragma unroll
  for (int ft = 0; ft < 8; ++ft)
#pragma unroll
    for (int v = 0; v < 8; ++v)
      Wh[(rowbase + lh * 8 + v) * GAT_F + ft * 16 + m] = acc[ft][v];
}

// ---------------------------------------------------------------------------
// Kernel 2: per row -> Wh f16 copy, s = Wh@a1, t = Wh@a2,
//           uw = (exp(t), exp(0.2t)), per-batch max(t). One block per batch.
// ---------------------------------------------------------------------------
__global__ __launch_bounds__(256) void gat_rows_kernel(
    const float* __restrict__ Wh, const float* __restrict__ a,
    _Float16* __restrict__ Whf16, float* __restrict__ s,
    float2* __restrict__ uw, float* __restrict__ tmax) {
  int b   = blockIdx.x;
  int tid = threadIdx.x;
  __shared__ float sa[2 * GAT_F];
  __shared__ float red[256];
  sa[tid] = a[tid];                 // 256 == 2*F
  __syncthreads();

  float lmax = -3.4e38f;
  for (int r = tid; r < GAT_N; r += 256) {
    long row = (long)b * GAT_N + r;
    const float* wr = Wh + row * GAT_F;
    _Float16*    wo = Whf16 + row * GAT_F;
    float d1 = 0.f, d2 = 0.f;
    for (int f = 0; f < GAT_F; f += 4) {
      float4 w4 = *(const float4*)(wr + f);
      d1 += w4.x * sa[f]     + w4.y * sa[f + 1] +
            w4.z * sa[f + 2] + w4.w * sa[f + 3];
      d2 += w4.x * sa[GAT_F + f]     + w4.y * sa[GAT_F + f + 1] +
            w4.z * sa[GAT_F + f + 2] + w4.w * sa[GAT_F + f + 3];
      wo[f]     = (_Float16)w4.x; wo[f + 1] = (_Float16)w4.y;
      wo[f + 2] = (_Float16)w4.z; wo[f + 3] = (_Float16)w4.w;
    }
    s[row] = d1;
    float2 p;
    p.x = __expf(d2);
    p.y = __expf(GAT_ALPHA * d2);
    uw[row] = p;
    lmax = fmaxf(lmax, d2);
  }
  red[tid] = lmax;
  __syncthreads();
  for (int off = 128; off > 0; off >>= 1) {
    if (tid < off) red[tid] = fmaxf(red[tid], red[tid + off]);
    __syncthreads();
  }
  if (tid == 0) tmax[b] = red[0];
}

// ---------------------------------------------------------------------------
// Kernel 3: fused softmax(lrelu(s_i + t_j)) @ Wh, then elu.
// Factorized probabilities (no transcendentals in the hot loop):
//   s_i+t_j > 0 : p = exp(s_i-m_i) * exp(t_j)      = A_i * u_j
//   else        : p = exp(0.2s_i-m_i)* exp(0.2t_j) = B_i * w_j
//   branch test:  u_j > exp(-s_i)  (exp is monotone)
// 4 waves / 64 rows per block; shared double-buffered 32x128 f16 Wh tile in
// LDS (padded stride 136 halfs: 16B-aligned ds_load_b128, 2-way conflicts max).
// Tile staging uses GLOBAL_LOAD_ASYNC_TO_LDS_B128 when available (ASYNCcnt);
// the async IOFFSET applies to both global and LDS addresses, matching our
// contiguous 64B/thread chunks.
// ---------------------------------------------------------------------------
__global__ __launch_bounds__(128) void gat_attn_kernel(
    const _Float16* __restrict__ Whf16, const float* __restrict__ s,
    const float2* __restrict__ uw, const float* __restrict__ tmaxp,
    float* __restrict__ out) {
  int b    = blockIdx.y;
  int tid  = threadIdx.x;
  int wave = tid >> 5;
  int lane = tid & 31;
  int m    = lane & 15;
  int lh   = lane >> 4;
  int k0   = lh * 8;
  int rowbase = blockIdx.x * 64 + wave * 16;

  // LDS: (u,w) table + double-buffered Wh tile (32 rows x 136-half stride)
  __shared__ __align__(16) float2 ulw[GAT_N];       // 32 KB
  __shared__ uint4 stage[2][32 * 17];               // 2 x 8704 B

  for (int i = tid; i < GAT_N / 2; i += 128) {      // 2048 float4 copies
    float4 v4 = ((const float4*)(uw + (long)b * GAT_N))[i];
    ((float4*)ulw)[i] = v4;
  }

  const _Float16* vbase = Whf16 + (long)b * GAT_N * GAT_F;
  int trow = tid >> 2;          // 0..31 : tile row
  int tcol = tid & 3;           // 0..3  : 32-half column chunk
  int sidx = trow * 17 + tcol * 4;

#if GAT_ASYNC_STAGE
  typedef __attribute__((address_space(1))) v4i* gvp_t;
  typedef __attribute__((address_space(3))) v4i* lvp_t;
  // prefetch tile 0 into stage[0] via async DMA
  {
    gvp_t g = (gvp_t)(vbase + (long)trow * GAT_F + tcol * 32);
    lvp_t l = (lvp_t)&stage[0][sidx];
    __builtin_amdgcn_global_load_async_to_lds_b128(g, l, 0, 0);
    __builtin_amdgcn_global_load_async_to_lds_b128(g, l, 16, 0);
    __builtin_amdgcn_global_load_async_to_lds_b128(g, l, 32, 0);
    __builtin_amdgcn_global_load_async_to_lds_b128(g, l, 48, 0);
  }
  gat_wait_async0();
#else
  uint4 pf0, pf1, pf2, pf3;
  {
    const uint4* gsrc = (const uint4*)(vbase + (long)trow * GAT_F + tcol * 32);
    pf0 = gsrc[0]; pf1 = gsrc[1]; pf2 = gsrc[2]; pf3 = gsrc[3];
  }
  stage[0][sidx] = pf0; stage[0][sidx + 1] = pf1;
  stage[0][sidx + 2] = pf2; stage[0][sidx + 3] = pf3;
#endif
  __syncthreads();

  float s_i = s[(long)b * GAT_N + rowbase + m];
  float e0  = s_i + tmaxp[b];
  float mi  = e0 > 0.f ? e0 : GAT_ALPHA * e0;       // exact row max of e
  float Af  = __expf(s_i - mi);
  float Bf  = __expf(GAT_ALPHA * s_i - mi);
  float es  = __expf(-s_i);

  v8f acc[8] = {};
  float zsum = 0.f;
  const int nb = GAT_N / 32;

  for (int it = 0; it < nb; ++it) {
    int cur = it & 1;
    bool have_next = (it + 1) < nb;
    if (have_next) {
#if GAT_ASYNC_STAGE
      gvp_t g = (gvp_t)(vbase + (long)((it + 1) * 32 + trow) * GAT_F + tcol * 32);
      lvp_t l = (lvp_t)&stage[cur ^ 1][sidx];
      __builtin_amdgcn_global_load_async_to_lds_b128(g, l, 0, 0);
      __builtin_amdgcn_global_load_async_to_lds_b128(g, l, 16, 0);
      __builtin_amdgcn_global_load_async_to_lds_b128(g, l, 32, 0);
      __builtin_amdgcn_global_load_async_to_lds_b128(g, l, 48, 0);
#else
      const uint4* gsrc =
          (const uint4*)(vbase + (long)((it + 1) * 32 + trow) * GAT_F + tcol * 32);
      pf0 = gsrc[0]; pf1 = gsrc[1]; pf2 = gsrc[2]; pf3 = gsrc[3];
#endif
    }

    int jb = it * 32;
    v16h af;
#pragma unroll
    for (int hh = 0; hh < 8; ++hh) {
      float2 uw1 = ulw[jb + k0 + hh];
      float p1 = (uw1.x > es) ? uw1.x * Af : uw1.y * Bf;
      zsum += p1;
      af[hh] = (_Float16)p1;
      float2 uw2 = ulw[jb + k0 + 16 + hh];
      float p2 = (uw2.x > es) ? uw2.x * Af : uw2.y * Bf;
      zsum += p2;
      af[8 + hh] = (_Float16)p2;
    }

#pragma unroll
    for (int ft = 0; ft < 8; ++ft) {
      const _Float16* sh = (const _Float16*)stage[cur] + k0 * 136 + ft * 16 + m;
      union { uint4 q[2]; v16h v; } bf;
#pragma unroll
      for (int hh = 0; hh < 8; ++hh) {
        bf.v[hh]     = sh[hh * 136];
        bf.v[8 + hh] = sh[(16 + hh) * 136];
      }
      acc[ft] = __builtin_amdgcn_wmma_f32_16x16x32_f16(
          false, af, false, bf.v, (short)0, acc[ft], false, false);
    }

#if GAT_ASYNC_STAGE
    gat_wait_async0();                 // staged writes visible before barrier
#else
    if (have_next) {
      stage[cur ^ 1][sidx] = pf0; stage[cur ^ 1][sidx + 1] = pf1;
      stage[cur ^ 1][sidx + 2] = pf2; stage[cur ^ 1][sidx + 3] = pf3;
    }
#endif
    __syncthreads();
  }

  // Z per row: lanes l and l+16 both carry row l%16 partials.
  float zo = zsum + __shfl_xor(zsum, 16, 32);       // lanes 0..15 hold Z_r

#pragma unroll
  for (int v = 0; v < 8; ++v) {
    int r = lh * 8 + v;
    float Zi = 1.0f / __shfl(zo, r, 32);            // 8 divides, not 64
#pragma unroll
    for (int ft = 0; ft < 8; ++ft) {
      float val = acc[ft][v] * Zi;
      val = val > 0.f ? val : (__expf(val) - 1.0f);   // elu
      out[((long)b * GAT_N + rowbase + r) * GAT_F + ft * 16 + m] = val;
    }
  }
}

// ---------------------------------------------------------------------------
extern "C" void kernel_launch(void* const* d_in, const int* in_sizes, int n_in,
                              void* d_out, int out_size, void* d_ws,
                              size_t ws_size, hipStream_t stream) {
  const float* h = (const float*)d_in[0];
  const float* W = (const float*)d_in[1];
  const float* a = (const float*)d_in[2];
  float* out = (float*)d_out;
  char* ws = (char*)d_ws;

  float*    Wh    = (float*)(ws + WS_WH_F32);
  _Float16* Whf16 = (_Float16*)(ws + WS_WH_F16);
  _Float16* Wf16  = (_Float16*)(ws + WS_W_F16);
  float*    sv    = (float*)(ws + WS_S);
  float2*   uwv   = (float2*)(ws + WS_UW);
  float*    tmx   = (float*)(ws + WS_TMAX);

  gat_wprep_kernel<<<(GAT_F * GAT_F + 255) / 256, 256, 0, stream>>>(W, Wf16);
  gat_gemm1_kernel<<<(GAT_B * GAT_N) / 64, 128, 0, stream>>>(h, Wf16, Wh);
  gat_rows_kernel<<<GAT_B, 256, 0, stream>>>(Wh, a, Whf16, sv, uwv, tmx);
  gat_attn_kernel<<<dim3(GAT_N / 64, GAT_B), 128, 0, stream>>>(Whf16, sv, uwv,
                                                               tmx, out);
}
